// EdgeAwareAttention_54168127537236
// MI455X (gfx1250) — matrix operand order repaired and checked by reading, not compiled, not run on hardware
//
#include <hip/hip_runtime.h>
#include <hip/hip_bf16.h>

// ---------------- problem constants ----------------
#define NN      10000           // nodes
#define NE      160000          // edges
#define HID     256             // hidden
#define NHEADS  8
#define HDIM    32              // head dim
#define MT      (NN / 16)       // 625 M-tiles
#define KT      (HID / 32)      // 8  K-tiles (K step 32 for bf16 WMMA)
#define NT      (HID / 16)      // 16 N-tiles
#define NBLK    4               // N-tiles per wave (16x64 strip)
#define LN_EPSF 1e-5f
#define SCALEF  0.17677669529663687f   // 1/sqrt(32)

typedef __attribute__((ext_vector_type(16))) __bf16 v16bf;
typedef __attribute__((ext_vector_type(8)))  float  v8f;

// float -> bf16 bits, round-to-nearest-even
__device__ __forceinline__ unsigned short f2bf(float f) {
    unsigned u = __float_as_uint(f);
    unsigned r = u + 0x7FFFu + ((u >> 16) & 1u);
    return (unsigned short)(r >> 16);
}

// order-preserving float<->uint for atomicMax on signed floats
__device__ __forceinline__ unsigned encF(float v) {
    unsigned b = __float_as_uint(v);
    return (b & 0x80000000u) ? ~b : (b | 0x80000000u);
}
__device__ __forceinline__ float decF(unsigned k) {
    unsigned b = (k & 0x80000000u) ? (k ^ 0x80000000u) : ~k;
    return __uint_as_float(b);
}

// ---------------- init scratch ----------------
__global__ void eaa_init_kernel(float* __restrict__ agg, float* __restrict__ denom,
                                unsigned* __restrict__ smaxKey) {
    int idx = blockIdx.x * blockDim.x + threadIdx.x;
    if (idx < NN * HID) agg[idx] = 0.0f;
    if (idx < NN * NHEADS) {
        denom[idx]   = 0.0f;
        smaxKey[idx] = 0x007FFFFFu;   // encF(-inf)
    }
}

// ---------------- repack X (f32 row-major) -> WMMA A-fragments (bf16) ----------------
// Apack layout: [mt][kt][lane(32)][elem(16)], elem->K mapping per CDNA5 ISA 16-bit A table
__global__ void eaa_pack_a_kernel(const float* __restrict__ X, unsigned short* __restrict__ out,
                                  int Mtiles) {
    int idx = blockIdx.x * blockDim.x + threadIdx.x;
    if (idx >= Mtiles * KT * 512) return;
    int i    = idx & 15;
    int lane = (idx >> 4) & 31;
    int tile = idx >> 9;            // mt*KT + kt
    int kt   = tile & (KT - 1);
    int mt   = tile >> 3;
    int row  = mt * 16 + (lane & 15);
    int half = lane >> 4;
    int kin  = (i < 8) ? (i + 8 * half) : (i + 8 + 8 * half);
    int k    = kt * 32 + kin;
    out[idx] = f2bf(X[(size_t)row * HID + k]);
}

// ---------------- repack W (f32 [K][N] row-major) -> WMMA B-fragments (bf16) ----------------
// Bpack layout: [nt][kt][lane(32)][elem(16)], elem i -> K = (i<8 ? i : i+8) + 8*(lane/16)
__global__ void eaa_pack_b_kernel(const float* __restrict__ W, unsigned short* __restrict__ out) {
    int idx = blockIdx.x * blockDim.x + threadIdx.x;
    if (idx >= NT * KT * 512) return;
    int i    = idx & 15;
    int lane = (idx >> 4) & 31;
    int tile = idx >> 9;            // nt*KT + kt
    int kt   = tile & (KT - 1);
    int nt   = tile >> 3;
    int col  = nt * 16 + (lane & 15);
    int k    = kt * 32 + ((i < 8) ? i : (i + 8)) + 8 * (lane >> 4);
    out[idx] = f2bf(W[(size_t)k * HID + col]);
}

// ---------------- bf16 WMMA GEMM: C[M,256] = A[M,256] x B[256,256] (+ residual) ----------------
// Wave computes a 16x64 strip: one A fragment feeds 4 WMMAs (N-blocking for A reuse).
__global__ void eaa_wmma_gemm_kernel(const unsigned short* __restrict__ Apack,
                                     const unsigned short* __restrict__ Bpack,
                                     const float* __restrict__ residual,
                                     float* __restrict__ C, int Mtiles) {
    int wid  = (blockIdx.x * blockDim.x + threadIdx.x) >> 5;   // wave-uniform
    int lane = threadIdx.x & 31;
    if (wid >= Mtiles * (NT / NBLK)) return;                   // whole wave exits together
    int mt = wid >> 2;            // / (NT/NBLK)
    int ng = wid & 3;             // N-group: columns [ng*64, ng*64+64)

    v8f acc0 = {}, acc1 = {}, acc2 = {}, acc3 = {};
    const unsigned short* aPtr = Apack + ((size_t)(mt * KT) * 32 + lane) * 16;
    const unsigned short* bPtr = Bpack + ((size_t)(ng * NBLK * KT) * 32 + lane) * 16;
    const size_t bStride = (size_t)KT * 512;   // one nt column of fragments
#pragma unroll
    for (int kt = 0; kt < KT; ++kt) {
        v16bf a  = *reinterpret_cast<const v16bf*>(aPtr + (size_t)kt * 512);
        v16bf b0 = *reinterpret_cast<const v16bf*>(bPtr + (size_t)kt * 512);
        v16bf b1 = *reinterpret_cast<const v16bf*>(bPtr + bStride     + (size_t)kt * 512);
        v16bf b2 = *reinterpret_cast<const v16bf*>(bPtr + bStride * 2 + (size_t)kt * 512);
        v16bf b3 = *reinterpret_cast<const v16bf*>(bPtr + bStride * 3 + (size_t)kt * 512);
        acc0 = __builtin_amdgcn_wmma_f32_16x16x32_bf16(false, a, false, b0, (short)0, acc0, false, false);
        acc1 = __builtin_amdgcn_wmma_f32_16x16x32_bf16(false, a, false, b1, (short)0, acc1, false, false);
        acc2 = __builtin_amdgcn_wmma_f32_16x16x32_bf16(false, a, false, b2, (short)0, acc2, false, false);
        acc3 = __builtin_amdgcn_wmma_f32_16x16x32_bf16(false, a, false, b3, (short)0, acc3, false, false);
    }

    // C/D fragment: VGPR r -> row = 8*(lane/16)+r, col = lane%16
    int half = lane >> 4, n = lane & 15;
    size_t rowBase = (size_t)(mt * 16 + half * 8) * HID;
    v8f accs[NBLK] = {acc0, acc1, acc2, acc3};
#pragma unroll
    for (int j = 0; j < NBLK; ++j) {
        size_t base = rowBase + (size_t)(ng * NBLK + j) * 16 + n;
        if (residual) {
#pragma unroll
            for (int r = 0; r < 8; ++r)
                C[base + (size_t)r * HID] = accs[j][r] + residual[base + (size_t)r * HID];
        } else {
#pragma unroll
            for (int r = 0; r < 8; ++r)
                C[base + (size_t)r * HID] = accs[j][r];
        }
    }
}

// ---------------- edge scores + segment max ----------------
__global__ void eaa_edge_scores_kernel(const float* __restrict__ Q, const float* __restrict__ K,
                                       const int* __restrict__ src, const int* __restrict__ dst,
                                       float* __restrict__ scores, unsigned* __restrict__ smaxKey) {
    int t = blockIdx.x * blockDim.x + threadIdx.x;
    if (t >= NE * NHEADS) return;
    int e = t >> 3, h = t & 7;
    int s = src[e], d = dst[e];
    const float4* q = reinterpret_cast<const float4*>(Q + (size_t)s * HID + h * HDIM);
    const float4* k = reinterpret_cast<const float4*>(K + (size_t)d * HID + h * HDIM);
    float acc = 0.0f;
#pragma unroll
    for (int j = 0; j < 8; ++j) {
        float4 qa = q[j], ka = k[j];
        acc += qa.x * ka.x + qa.y * ka.y + qa.z * ka.z + qa.w * ka.w;
    }
    acc *= SCALEF;
    scores[t] = acc;
    atomicMax(&smaxKey[s * NHEADS + h], encF(acc));
}

// ---------------- segment sum of exp(s - max) ----------------
__global__ void eaa_edge_denom_kernel(const float* __restrict__ scores, const int* __restrict__ src,
                                      const unsigned* __restrict__ smaxKey,
                                      float* __restrict__ denom) {
    int t = blockIdx.x * blockDim.x + threadIdx.x;
    if (t >= NE * NHEADS) return;
    int e = t >> 3, h = t & 7;
    int s = src[e];
    float m = decF(smaxKey[s * NHEADS + h]);
    atomicAdd(&denom[s * NHEADS + h], expf(scores[t] - m));
}

// ---------------- weighted V aggregation ----------------
__global__ void eaa_edge_agg_kernel(const float* __restrict__ scores, const float* __restrict__ V,
                                    const int* __restrict__ src, const int* __restrict__ dst,
                                    const unsigned* __restrict__ smaxKey,
                                    const float* __restrict__ denom, float* __restrict__ agg) {
    int t = blockIdx.x * blockDim.x + threadIdx.x;
    if (t >= NE * NHEADS) return;
    int e = t >> 3, h = t & 7;
    int s = src[e], d = dst[e];
    float w = expf(scores[t] - decF(smaxKey[s * NHEADS + h])) / denom[s * NHEADS + h];
    const float* vp = V + (size_t)d * HID + h * HDIM;
    float* ap = agg + (size_t)s * HID + h * HDIM;
#pragma unroll
    for (int j = 0; j < HDIM; ++j) atomicAdd(&ap[j], w * vp[j]);
}

// ---------------- wave-per-node LayerNorm ----------------
__global__ void eaa_layernorm_kernel(const float* __restrict__ y, const float* __restrict__ gamma,
                                     const float* __restrict__ beta, float* __restrict__ out) {
    int wid  = (blockIdx.x * blockDim.x + threadIdx.x) >> 5;
    int lane = threadIdx.x & 31;
    if (wid >= NN) return;
    const float4* row = reinterpret_cast<const float4*>(y + (size_t)wid * HID);
    float4 a = row[lane];
    float4 b = row[lane + 32];
    float s  = a.x + a.y + a.z + a.w + b.x + b.y + b.z + b.w;
    float s2 = a.x * a.x + a.y * a.y + a.z * a.z + a.w * a.w +
               b.x * b.x + b.y * b.y + b.z * b.z + b.w * b.w;
#pragma unroll
    for (int off = 16; off > 0; off >>= 1) {
        s  += __shfl_xor(s,  off, 32);
        s2 += __shfl_xor(s2, off, 32);
    }
    float mu   = s * (1.0f / HID);
    float var  = s2 * (1.0f / HID) - mu * mu;
    float rstd = rsqrtf(var + LN_EPSF);
    const float4* g4 = reinterpret_cast<const float4*>(gamma);
    const float4* b4 = reinterpret_cast<const float4*>(beta);
    float4* o = reinterpret_cast<float4*>(out + (size_t)wid * HID);
    float4 g = g4[lane], be = b4[lane];
    float4 r;
    r.x = (a.x - mu) * rstd * g.x + be.x;  r.y = (a.y - mu) * rstd * g.y + be.y;
    r.z = (a.z - mu) * rstd * g.z + be.z;  r.w = (a.w - mu) * rstd * g.w + be.w;
    o[lane] = r;
    g = g4[lane + 32]; be = b4[lane + 32];
    r.x = (b.x - mu) * rstd * g.x + be.x;  r.y = (b.y - mu) * rstd * g.y + be.y;
    r.z = (b.z - mu) * rstd * g.z + be.z;  r.w = (b.w - mu) * rstd * g.w + be.w;
    o[lane + 32] = r;
}

// ---------------- host side ----------------
extern "C" void kernel_launch(void* const* d_in, const int* in_sizes, int n_in,
                              void* d_out, int out_size, void* d_ws, size_t ws_size,
                              hipStream_t stream) {
    (void)in_sizes; (void)n_in; (void)out_size; (void)ws_size;

    const float* X     = (const float*)d_in[0];          // (1,10000,256)
    const int*   eidx  = (const int*)d_in[1];            // (2,160000)
    const float* Wq    = (const float*)d_in[2];
    const float* Wk    = (const float*)d_in[3];
    const float* Wv    = (const float*)d_in[4];
    const float* Wo    = (const float*)d_in[5];
    const float* gamma = (const float*)d_in[6];
    const float* beta  = (const float*)d_in[7];
    float*       out   = (float*)d_out;
    const int* src = eidx;
    const int* dst = eidx + NE;

    // workspace carve-up (256B aligned)
    char* ws = (char*)d_ws;
    size_t off = 0;
    auto carve = [&](size_t bytes) -> void* {
        void* p = ws + off;
        off = (off + bytes + 255) & ~(size_t)255;
        return p;
    };
    unsigned short* Xpack   = (unsigned short*)carve((size_t)NN * HID * 2);
    unsigned short* WqPack  = (unsigned short*)carve((size_t)HID * HID * 2);
    unsigned short* WkPack  = (unsigned short*)carve((size_t)HID * HID * 2);
    unsigned short* WvPack  = (unsigned short*)carve((size_t)HID * HID * 2);
    unsigned short* WoPack  = (unsigned short*)carve((size_t)HID * HID * 2);
    float*          Q       = (float*)carve((size_t)NN * HID * 4);
    float*          Km      = (float*)carve((size_t)NN * HID * 4);
    float*          V       = (float*)carve((size_t)NN * HID * 4);
    float*          scores  = (float*)carve((size_t)NE * NHEADS * 4);
    unsigned*       smaxKey = (unsigned*)carve((size_t)NN * NHEADS * 4);
    float*          denom   = (float*)carve((size_t)NN * NHEADS * 4);
    float*          agg     = (float*)carve((size_t)NN * HID * 4);
    unsigned short* aggPack = (unsigned short*)carve((size_t)NN * HID * 2);
    float*          y       = (float*)carve((size_t)NN * HID * 4);

    const int B = 256;
    // 1) init scratch (agg/denom/smax re-inited every call -> deterministic work)
    eaa_init_kernel<<<(NN * HID + B - 1) / B, B, 0, stream>>>(agg, denom, smaxKey);

    // 2) repack operands into WMMA fragment layouts (bf16)
    eaa_pack_a_kernel<<<(MT * KT * 512 + B - 1) / B, B, 0, stream>>>(X, Xpack, MT);
    eaa_pack_b_kernel<<<(NT * KT * 512 + B - 1) / B, B, 0, stream>>>(Wq, WqPack);
    eaa_pack_b_kernel<<<(NT * KT * 512 + B - 1) / B, B, 0, stream>>>(Wk, WkPack);
    eaa_pack_b_kernel<<<(NT * KT * 512 + B - 1) / B, B, 0, stream>>>(Wv, WvPack);
    eaa_pack_b_kernel<<<(NT * KT * 512 + B - 1) / B, B, 0, stream>>>(Wo, WoPack);

    // 3) Q/K/V projections on the WMMA pipe (wave = 16x64 strip, A reused 4x)
    int gemmWaves  = MT * (NT / NBLK);
    int gemmBlocks = (gemmWaves * 32 + B - 1) / B;
    eaa_wmma_gemm_kernel<<<gemmBlocks, B, 0, stream>>>(Xpack, WqPack, nullptr, Q,  MT);
    eaa_wmma_gemm_kernel<<<gemmBlocks, B, 0, stream>>>(Xpack, WkPack, nullptr, Km, MT);
    eaa_wmma_gemm_kernel<<<gemmBlocks, B, 0, stream>>>(Xpack, WvPack, nullptr, V,  MT);

    // 4-6) edge-softmax aggregation
    int edgeBlocks = (NE * NHEADS + B - 1) / B;
    eaa_edge_scores_kernel<<<edgeBlocks, B, 0, stream>>>(Q, Km, src, dst, scores, smaxKey);
    eaa_edge_denom_kernel<<<edgeBlocks, B, 0, stream>>>(scores, src, smaxKey, denom);
    eaa_edge_agg_kernel<<<edgeBlocks, B, 0, stream>>>(scores, V, src, dst, smaxKey, denom, agg);

    // 7) output projection + residual on the WMMA pipe
    eaa_pack_a_kernel<<<(MT * KT * 512 + B - 1) / B, B, 0, stream>>>(agg, aggPack, MT);
    eaa_wmma_gemm_kernel<<<gemmBlocks, B, 0, stream>>>(aggPack, WoPack, X, y, MT);

    // 8) LayerNorm
    eaa_layernorm_kernel<<<(NN * 32 + B - 1) / B, B, 0, stream>>>(y, gamma, beta, out);
}
